// TimeMixing_11828339933837
// MI455X (gfx1250) — matrix operand order repaired and checked by reading, not compiled, not run on hardware
//
#include <hip/hip_runtime.h>
#include <hip/hip_bf16.h>
#include <cstdint>
#include <cstddef>

// ---------------------------------------------------------------------------
// RWKV6 time-mixing for MI455X (gfx1250, wave32, WMMA).
// GEMMs: v_wmma_f32_16x16x32_bf16 with async-to-LDS double-buffered B tiles.
// Sequential wkv6 scan on VALU (registers + LDS broadcast).
// ---------------------------------------------------------------------------

typedef __attribute__((ext_vector_type(16))) __bf16 v16bf;
typedef __attribute__((ext_vector_type(8)))  __bf16 v8bf;
typedef __attribute__((ext_vector_type(8)))  float  v8f;

#define BD 8
#define TD 2048
#define CD 2048
#define AD 2048
#define HD 32
#define ND 64
#define MD (BD*TD)          // 16384 rows
#define LORA_MIX 32
#define LORA_DEC 64
#define HMIX_PAD 192        // 5*32=160 padded to multiple of 64

__device__ __forceinline__ __bf16 to_bf16(float f) { return (__bf16)f; }

// --------------------------- elementwise prep ------------------------------

__global__ void __launch_bounds__(256)
cvt_bf16_kernel(const float* __restrict__ in, __bf16* __restrict__ out, size_t n) {
  size_t i = (size_t)blockIdx.x * blockDim.x + threadIdx.x;
  if (i < n) out[i] = to_bf16(in[i]);
}

// in: (R, Cc) row-major f32  ->  out: (Cpad, R) row-major bf16, zero-padded rows
__global__ void __launch_bounds__(256)
transpose_cvt_kernel(const float* __restrict__ in, __bf16* __restrict__ out,
                     int R, int Cc, int Cpad) {
  size_t i = (size_t)blockIdx.x * blockDim.x + threadIdx.x;
  size_t total = (size_t)Cpad * R;
  if (i >= total) return;
  int r  = (int)(i % R);
  int oc = (int)(i / R);
  out[i] = (oc < Cc) ? to_bf16(in[(size_t)r * Cc + oc]) : to_bf16(0.f);
}

// token shift: xx = x[t-1]-x[t] (0 at t=0); xxx = bf16(x + xx*time_maa_x)
__global__ void __launch_bounds__(256)
shift_kernel(const float* __restrict__ x, const float* __restrict__ tmx,
             float* __restrict__ xx, __bf16* __restrict__ xxx) {
  size_t idx = (size_t)blockIdx.x * blockDim.x + threadIdx.x;
  if (idx >= (size_t)MD * CD) return;
  int c = (int)(idx % CD);
  int t = (int)((idx / CD) % TD);
  float xv   = x[idx];
  float prev = (t > 0) ? x[idx - CD] : 0.f;
  float d    = prev - xv;
  xx[idx]  = d;
  xxx[idx] = to_bf16(xv + d * tmx[c]);
}

// --------------------------- WMMA bf16 GEMM --------------------------------
// Out[M,N] = X[M,K] * W[N,K]^T   (both operands row-major along K)
// WG 256 threads = 8 waves; each wave owns a 32x64 stripe (8 WMMAs / k-step).
// All waves share one 64-col group: the 64x32 bf16 B tile (4KB) is staged in
// LDS with double-buffered global_load_async_to_lds_b128 (ASYNCcnt).

enum { E_NONE = 0, E_SILU = 1, E_DECAY = 2, E_TANH_BF16 = 3 };

template <int EPI>
__global__ void __launch_bounds__(256)
gemm_bf16_wmma(const __bf16* __restrict__ X, const __bf16* __restrict__ W,
               float* __restrict__ outF, __bf16* __restrict__ outB,
               const float* __restrict__ bias, int Mm, int Nn, int Kk) {
  __shared__ __align__(16) __bf16 Bs[2][64 * 32];   // 2 x 4KB B tiles
  const int tid  = threadIdx.x;
  const int lane = tid & 31;
  const int wv   = tid >> 5;
  const int lh   = lane >> 4;        // lane half (0/1)
  const int ll   = lane & 15;
  const int mrow0 = (blockIdx.x * 8 + wv) * 32;     // 32 rows per wave
  const int nbase = blockIdx.y * 64;

  // Cooperative async fill: 256 threads x 16B == one 64x32 bf16 tile.
  const int fn = tid >> 2;                 // local B row 0..63
  const int fk = (tid & 3) * 8;            // bf16 offset 0/8/16/24 within k-step
  const __bf16* gB = W + (size_t)(nbase + fn) * Kk + fk;
  // low 32 bits of a generic LDS pointer == LDS byte address
  const unsigned ldsB0 =
      (unsigned)(size_t)(&Bs[0][0]) + (unsigned)(fn * 64 + (tid & 3) * 16);

  // A-fragment (16x32 bf16): lanes 0-15 row M=ll hold K[0..15],
  // lanes 16-31 hold K[16..31] -> two contiguous 16B loads.
  const __bf16* ap0 = X + (size_t)(mrow0 + ll) * Kk + (lh ? 16 : 0);
  const __bf16* ap1 = ap0 + (size_t)16 * Kk;

  v8f z = {};
  v8f acc[8] = {z, z, z, z, z, z, z, z};

  // prologue: async fill of buffer 0 with k0 = 0
  asm volatile("global_load_async_to_lds_b128 %0, %1, off"
               :: "v"(ldsB0), "v"(gB) : "memory");

  for (int k0 = 0; k0 < Kk; k0 += 32) {
    const int cur = (k0 >> 5) & 1;
    if (k0 + 32 < Kk) {
      // fill the other buffer with the next k-step's tile
      unsigned dst = ldsB0 + (cur ? 0u : 4096u);
      const __bf16* src = gB + k0 + 32;
      asm volatile("global_load_async_to_lds_b128 %0, %1, off"
                   :: "v"(dst), "v"(src) : "memory");
      // two fills outstanding per wave; oldest (for `cur`) done at cnt<=1
      asm volatile("s_wait_asynccnt 1" ::: "memory");
    } else {
      asm volatile("s_wait_asynccnt 0" ::: "memory");
    }
    __syncthreads();

    v8bf alo0 = *(const v8bf*)(ap0 + k0);
    v8bf ahi0 = *(const v8bf*)(ap0 + k0 + 8);
    v8bf alo1 = *(const v8bf*)(ap1 + k0);
    v8bf ahi1 = *(const v8bf*)(ap1 + k0 + 8);
    v16bf a0, a1;
#pragma unroll
    for (int e = 0; e < 8; ++e) {
      a0[e] = alo0[e]; a0[e + 8] = ahi0[e];
      a1[e] = alo1[e]; a1[e + 8] = ahi1[e];
    }
    __builtin_prefetch(ap0 + k0 + 64, 0, 1);   // global_prefetch
    __builtin_prefetch(ap1 + k0 + 64, 0, 1);

    // B-fragment (32x16 bf16): lane reads 16 contiguous bf16 of row N=ll at
    // K + (lane>=16 ? 16 : 0) from the LDS tile.
#pragma unroll
    for (int tt = 0; tt < 4; ++tt) {
      const v16bf bfrag =
          *(const v16bf*)(&Bs[cur][(tt * 16 + ll) * 32 + lh * 16]);
      acc[tt] = __builtin_amdgcn_wmma_f32_16x16x32_bf16(
          false, a0, false, bfrag, (short)0, acc[tt], false, false);
      acc[tt + 4] = __builtin_amdgcn_wmma_f32_16x16x32_bf16(
          false, a1, false, bfrag, (short)0, acc[tt + 4], false, false);
    }
    __syncthreads();   // nobody overwrites `cur` until all waves finished it
  }

  // C/D layout: lane half selects M+8; N = ll within tile.
#pragma unroll
  for (int tt = 0; tt < 4; ++tt) {
    int n = nbase + tt * 16 + ll;
#pragma unroll
    for (int half = 0; half < 2; ++half) {
      v8f a = acc[tt + half * 4];
#pragma unroll
      for (int i = 0; i < 8; ++i) {
        int m = mrow0 + half * 16 + i + lh * 8;
        size_t o = (size_t)m * Nn + n;
        float v = a[i];
        if constexpr (EPI == E_NONE) {
          outF[o] = v;
        } else if constexpr (EPI == E_SILU) {
          outF[o] = v / (1.f + __expf(-v));
        } else if constexpr (EPI == E_DECAY) {
          float w = bias[n] + v;
          outF[o] = __expf(-__expf(w));
        } else {  // E_TANH_BF16
          outB[o] = to_bf16(tanhf(v));
        }
      }
    }
  }
}

// ---------------- fused second LoRA stage + 5-way mix (K=32, VALU) ---------
// m_f[c] = sum_j tanh_h[f*32+j] * maa_w2[f,j,c];  xq = x + xx*(tm_q + m_q)

__global__ void __launch_bounds__(256)
mix_kernel(const float* __restrict__ x, const float* __restrict__ xx,
           const __bf16* __restrict__ hmix, const float* __restrict__ maa_w2,
           const float* __restrict__ tmw, const float* __restrict__ tmk,
           const float* __restrict__ tmv, const float* __restrict__ tmr,
           const float* __restrict__ tmg,
           __bf16* __restrict__ xw, __bf16* __restrict__ xk,
           __bf16* __restrict__ xvp, __bf16* __restrict__ xr,
           __bf16* __restrict__ xg) {
  __shared__ float h[5 * LORA_MIX];
  int row = blockIdx.x;
  int tid = threadIdx.x;
  if (tid < 5 * LORA_MIX) h[tid] = (float)hmix[(size_t)row * HMIX_PAD + tid];
  __syncthreads();
  const size_t rb = (size_t)row * CD;
#pragma unroll
  for (int j = 0; j < CD / 256; ++j) {
    int c = tid + j * 256;
    float m[5];
#pragma unroll
    for (int f = 0; f < 5; ++f) {
      float s = 0.f;
#pragma unroll
      for (int jj = 0; jj < LORA_MIX; ++jj)
        s += h[f * LORA_MIX + jj] * maa_w2[(size_t)(f * LORA_MIX + jj) * CD + c];
      m[f] = s;
    }
    float x0 = x[rb + c], d = xx[rb + c];
    xw[rb + c]  = to_bf16(x0 + d * (tmw[c] + m[0]));
    xk[rb + c]  = to_bf16(x0 + d * (tmk[c] + m[1]));
    xvp[rb + c] = to_bf16(x0 + d * (tmv[c] + m[2]));
    xr[rb + c]  = to_bf16(x0 + d * (tmr[c] + m[3]));
    xg[rb + c]  = to_bf16(x0 + d * (tmg[c] + m[4]));
  }
}

// --------------------------- wkv6 recurrent scan ---------------------------
// One block per (b,h). Thread j owns S[:,j] in 64 registers.

__global__ void __launch_bounds__(64)
wkv6_kernel(const float* __restrict__ r, const float* __restrict__ k,
            const float* __restrict__ v, const float* __restrict__ dec,
            const float* __restrict__ u, float* __restrict__ y) {
  int bh = blockIdx.x;
  int b = bh / HD, hh = bh % HD;
  int j = threadIdx.x;
  __shared__ float rL[ND], kL[ND], dL[ND], uL[ND];
  uL[j] = u[hh * ND + j];
  float S[ND];
#pragma unroll
  for (int i = 0; i < ND; ++i) S[i] = 0.f;
  __syncthreads();
  for (int t = 0; t < TD; ++t) {
    size_t base = ((size_t)b * TD + t) * AD + hh * ND;
    rL[j] = r[base + j];
    kL[j] = k[base + j];
    dL[j] = dec[base + j];
    float vj = v[base + j];
    __syncthreads();
    float yj = 0.f;
#pragma unroll
    for (int i = 0; i < ND; ++i) {
      float a = kL[i] * vj;
      yj += rL[i] * (S[i] + uL[i] * a);
      S[i] = dL[i] * S[i] + a;
    }
    y[base + j] = yj;
    __syncthreads();
  }
}

// --------------------- per-head GroupNorm + SiLU gate ----------------------

__global__ void __launch_bounds__(256)
gn_gate_kernel(const float* __restrict__ y, const float* __restrict__ g,
               const float* __restrict__ lnw, const float* __restrict__ lnb,
               __bf16* __restrict__ out) {
  __shared__ float s1[256], s2[256];
  int row = blockIdx.x, tid = threadIdx.x;
  size_t rb = (size_t)row * CD;
  int c0 = tid * 8;                 // 8 contiguous cols; 8 threads per head
  float vals[8];
  float su = 0.f, sq = 0.f;
#pragma unroll
  for (int e = 0; e < 8; ++e) {
    float vv = y[rb + c0 + e];
    vals[e] = vv; su += vv; sq += vv * vv;
  }
  s1[tid] = su; s2[tid] = sq;
  __syncthreads();
  int hb = (tid >> 3) << 3;
  float hs = 0.f, hq = 0.f;
#pragma unroll
  for (int e = 0; e < 8; ++e) { hs += s1[hb + e]; hq += s2[hb + e]; }
  float mu  = hs * (1.f / 64.f);
  float var = hq * (1.f / 64.f) - mu * mu;
  float rs  = rsqrtf(var + 6.4e-4f);   // EPS = 1e-5 * 8^2
#pragma unroll
  for (int e = 0; e < 8; ++e) {
    int c = c0 + e;
    float yn = (vals[e] - mu) * rs * lnw[c] + lnb[c];
    out[rb + c] = to_bf16(yn * g[rb + c]);
  }
}

// ------------------------------- launcher ----------------------------------

extern "C" void kernel_launch(void* const* d_in, const int* in_sizes, int n_in,
                              void* d_out, int out_size, void* d_ws, size_t ws_size,
                              hipStream_t stream) {
  (void)in_sizes; (void)n_in; (void)out_size; (void)ws_size;
  const float* x        = (const float*)d_in[0];
  const float* tmx      = (const float*)d_in[1];
  const float* tmw      = (const float*)d_in[2];
  const float* tmk      = (const float*)d_in[3];
  const float* tmv      = (const float*)d_in[4];
  const float* tmr      = (const float*)d_in[5];
  const float* tmg      = (const float*)d_in[6];
  const float* maa_w1   = (const float*)d_in[7];    // (C, 160)
  const float* maa_w2   = (const float*)d_in[8];    // (5, 32, C)
  const float* tdecay   = (const float*)d_in[9];    // (A)
  const float* decay_w1 = (const float*)d_in[10];   // (C, 64)
  const float* decay_w2 = (const float*)d_in[11];   // (64, A)
  const float* u        = (const float*)d_in[12];   // (H, N)
  const float* Wr       = (const float*)d_in[13];   // (A, C)
  const float* Wk       = (const float*)d_in[14];
  const float* Wv       = (const float*)d_in[15];
  const float* Wg       = (const float*)d_in[16];
  const float* Wo       = (const float*)d_in[17];   // (C, A)
  const float* lnw      = (const float*)d_in[18];
  const float* lnb      = (const float*)d_in[19];
  float* out            = (float*)d_out;

  char* ws = (char*)d_ws;
  size_t off = 0;
  auto alloc = [&](size_t bytes) {
    size_t o = off; off += (bytes + 255) & ~(size_t)255; return o;
  };
  const size_t MC4 = (size_t)MD * CD * 4;
  const size_t MC2 = (size_t)MD * CD * 2;

  size_t o_xx   = alloc(MC4);                       // xx f32  (reused as y)
  size_t o_xxx  = alloc(MC2);                       // xxx bf16 (reused as ybf)
  size_t o_hmx  = alloc((size_t)MD * HMIX_PAD * 2); // tanh(mix LoRA) bf16
  size_t o_xw   = alloc(MC2);
  size_t o_xk   = alloc(MC2);
  size_t o_xv   = alloc(MC2);
  size_t o_xr   = alloc(MC2);
  size_t o_xg   = alloc(MC2);
  size_t o_h2   = alloc((size_t)MD * LORA_DEC * 2); // tanh(decay LoRA) bf16
  size_t o_dec  = alloc(MC4);
  size_t o_R    = alloc(MC4);
  size_t o_K    = alloc(MC4);
  size_t o_V    = alloc(MC4);
  size_t o_g    = alloc(MC4);
  size_t o_w1m  = alloc((size_t)HMIX_PAD * CD * 2); // maa_w1^T padded (192,2048)
  size_t o_w1d  = alloc((size_t)LORA_DEC * CD * 2); // decay_w1^T (64,2048)
  size_t o_w2d  = alloc((size_t)AD * LORA_DEC * 2); // decay_w2^T (2048,64)
  size_t o_Wr   = alloc((size_t)AD * CD * 2);
  size_t o_Wk   = alloc((size_t)AD * CD * 2);
  size_t o_Wv   = alloc((size_t)AD * CD * 2);
  size_t o_Wg   = alloc((size_t)AD * CD * 2);
  size_t o_Wo   = alloc((size_t)CD * AD * 2);

  float*  xx   = (float*)(ws + o_xx);
  __bf16* xxx  = (__bf16*)(ws + o_xxx);
  __bf16* hmx  = (__bf16*)(ws + o_hmx);
  __bf16* xw   = (__bf16*)(ws + o_xw);
  __bf16* xk   = (__bf16*)(ws + o_xk);
  __bf16* xv   = (__bf16*)(ws + o_xv);
  __bf16* xr   = (__bf16*)(ws + o_xr);
  __bf16* xg   = (__bf16*)(ws + o_xg);
  __bf16* h2   = (__bf16*)(ws + o_h2);
  float*  dec  = (float*)(ws + o_dec);
  float*  Rb   = (float*)(ws + o_R);
  float*  Kb   = (float*)(ws + o_K);
  float*  Vb   = (float*)(ws + o_V);
  float*  gb   = (float*)(ws + o_g);
  __bf16* w1m  = (__bf16*)(ws + o_w1m);
  __bf16* w1d  = (__bf16*)(ws + o_w1d);
  __bf16* w2d  = (__bf16*)(ws + o_w2d);
  __bf16* Wrb  = (__bf16*)(ws + o_Wr);
  __bf16* Wkb  = (__bf16*)(ws + o_Wk);
  __bf16* Wvb  = (__bf16*)(ws + o_Wv);
  __bf16* Wgb  = (__bf16*)(ws + o_Wg);
  __bf16* Wob  = (__bf16*)(ws + o_Wo);
  float*  y    = (float*)(ws + o_xx);   // alias: xx dead after mix_kernel
  __bf16* ybf  = (__bf16*)(ws + o_xxx); // alias: xxx dead after mix-LoRA GEMM

  auto blocks = [](size_t n) { return dim3((unsigned)((n + 255) / 256)); };

  // ---- weight prep ----
  size_t wn = (size_t)AD * CD;
  cvt_bf16_kernel<<<blocks(wn), 256, 0, stream>>>(Wr, Wrb, wn);
  cvt_bf16_kernel<<<blocks(wn), 256, 0, stream>>>(Wk, Wkb, wn);
  cvt_bf16_kernel<<<blocks(wn), 256, 0, stream>>>(Wv, Wvb, wn);
  cvt_bf16_kernel<<<blocks(wn), 256, 0, stream>>>(Wg, Wgb, wn);
  cvt_bf16_kernel<<<blocks(wn), 256, 0, stream>>>(Wo, Wob, wn);
  transpose_cvt_kernel<<<blocks((size_t)HMIX_PAD * CD), 256, 0, stream>>>(
      maa_w1, w1m, CD, 5 * LORA_MIX, HMIX_PAD);
  transpose_cvt_kernel<<<blocks((size_t)LORA_DEC * CD), 256, 0, stream>>>(
      decay_w1, w1d, CD, LORA_DEC, LORA_DEC);
  transpose_cvt_kernel<<<blocks((size_t)AD * LORA_DEC), 256, 0, stream>>>(
      decay_w2, w2d, LORA_DEC, AD, AD);

  // ---- token shift ----
  shift_kernel<<<blocks((size_t)MD * CD), 256, 0, stream>>>(x, tmx, xx, xxx);

  // ---- mix LoRA stage 1: tanh(xxx @ maa_w1), N padded to 192 ----
  gemm_bf16_wmma<E_TANH_BF16><<<dim3(MD / 256, HMIX_PAD / 64), 256, 0, stream>>>(
      xxx, w1m, nullptr, hmx, nullptr, MD, HMIX_PAD, CD);

  // ---- fused LoRA stage 2 + 5-way mix ----
  mix_kernel<<<dim3(MD), 256, 0, stream>>>(x, xx, hmx, maa_w2,
                                           tmw, tmk, tmv, tmr, tmg,
                                           xw, xk, xv, xr, xg);

  // ---- decay: tanh(xw @ decay_w1), then exp(-exp(td + h2 @ decay_w2)) ----
  gemm_bf16_wmma<E_TANH_BF16><<<dim3(MD / 256, LORA_DEC / 64), 256, 0, stream>>>(
      xw, w1d, nullptr, h2, nullptr, MD, LORA_DEC, CD);
  gemm_bf16_wmma<E_DECAY><<<dim3(MD / 256, AD / 64), 256, 0, stream>>>(
      h2, w2d, dec, nullptr, tdecay, MD, AD, LORA_DEC);

  // ---- projections ----
  gemm_bf16_wmma<E_NONE><<<dim3(MD / 256, AD / 64), 256, 0, stream>>>(
      xr, Wrb, Rb, nullptr, nullptr, MD, AD, CD);
  gemm_bf16_wmma<E_NONE><<<dim3(MD / 256, AD / 64), 256, 0, stream>>>(
      xk, Wkb, Kb, nullptr, nullptr, MD, AD, CD);
  gemm_bf16_wmma<E_NONE><<<dim3(MD / 256, AD / 64), 256, 0, stream>>>(
      xv, Wvb, Vb, nullptr, nullptr, MD, AD, CD);
  gemm_bf16_wmma<E_SILU><<<dim3(MD / 256, AD / 64), 256, 0, stream>>>(
      xg, Wgb, gb, nullptr, nullptr, MD, AD, CD);

  // ---- wkv6 scan ----
  wkv6_kernel<<<dim3(BD * HD), 64, 0, stream>>>(Rb, Kb, Vb, dec, u, y);

  // ---- GroupNorm + gate -> bf16 ----
  gn_gate_kernel<<<dim3(MD), 256, 0, stream>>>(y, gb, lnw, lnb, ybf);

  // ---- output projection ----
  gemm_bf16_wmma<E_NONE><<<dim3(MD / 256, CD / 64), 256, 0, stream>>>(
      ybf, Wob, out, nullptr, nullptr, MD, CD, AD);
}
